// Multihead_attention_36232344109754
// MI455X (gfx1250) — matrix operand order repaired and checked by reading, not compiled, not run on hardware
//
#include <hip/hip_runtime.h>
#include <hip/hip_bf16.h>
#include <cstdint>

// ---------------------------------------------------------------------------
// CDNA5 (gfx1250) multi-head attention, f16 WMMA path, f32 accumulate.
// LDS-staged tiles (async-copy via GLOBAL_LOAD_ASYNC_TO_LDS_B128),
// 32x64 GEMM wave tiles.
// ---------------------------------------------------------------------------

typedef __attribute__((ext_vector_type(16))) _Float16 v16h;
typedef __attribute__((ext_vector_type(8)))  _Float16 v8h;
typedef __attribute__((ext_vector_type(8)))  float    v8f;
typedef __attribute__((ext_vector_type(4)))  int      v4i;

// address-space-qualified pointers for the async builtin
typedef __attribute__((address_space(1))) v4i* gv4i_p;   // global
typedef __attribute__((address_space(3))) v4i* lv4i_p;   // LDS

union V16 { v16h v; v8h h[2]; };

constexpr int BB = 4;          // batch
constexpr int S  = 1024;       // sequence
constexpr int D  = 1024;       // model dim
constexpr int H  = 16;         // heads
constexpr int DH = 64;         // head dim
constexpr int MM = BB * S;     // 4096 rows

static __device__ __forceinline__ v8f wmma_f16(v16h a, v16h b, v8f c) {
  return __builtin_amdgcn_wmma_f32_16x16x32_f16(
      /*neg_a=*/false, a, /*neg_b=*/false, b,
      /*c_mod=*/(short)0, c, /*reuse_a=*/false, /*reuse_b=*/false);
}

// ---- CDNA5 async global->LDS copy (GLOBAL_LOAD_ASYNC_TO_LDS_B128), guarded ----
#if defined(__has_builtin)
#if __has_builtin(__builtin_amdgcn_global_load_async_to_lds_b128) && \
    __has_builtin(__builtin_amdgcn_s_wait_asynccnt)
#define HAVE_ASYNC_LDS 1
#endif
#endif
#ifndef HAVE_ASYNC_LDS
#define HAVE_ASYNC_LDS 0
#endif

static __device__ __forceinline__ void stage_b128(const _Float16* g, _Float16* l) {
#if HAVE_ASYNC_LDS
  __builtin_amdgcn_global_load_async_to_lds_b128(
      (gv4i_p)(uintptr_t)g,
      (lv4i_p)(uint32_t)(uintptr_t)l,
      /*offset=*/0, /*cpol=*/0);
#else
  *(v8h*)l = *(const v8h*)g;
#endif
}

static __device__ __forceinline__ void stage_wait() {
#if HAVE_ASYNC_LDS
  __builtin_amdgcn_s_wait_asynccnt(0);   // own copies done before barrier signal
#endif
}

// ---------------------------------------------------------------------------
// 1a) Weight convert + transpose: Wt[n*D + k] = (f16) W[k*D + n]
// ---------------------------------------------------------------------------
__global__ void wconv_kernel(const float* __restrict__ W, _Float16* __restrict__ Wt) {
  int idx = blockIdx.x * blockDim.x + threadIdx.x;   // over D*D
  int k = idx >> 10;
  int n = idx & (D - 1);
  Wt[(size_t)n * D + k] = (_Float16)W[idx];
}

// 1b) Activation convert (flat, coalesced): Xh = (f16) X
__global__ void aconv_kernel(const float* __restrict__ X, _Float16* __restrict__ Xh) {
  int idx = blockIdx.x * blockDim.x + threadIdx.x;
  Xh[idx] = (_Float16)X[idx];
}

// ---------------------------------------------------------------------------
// 2) GEMM: C[M,N] = A[M,K](f16) @ Wt^T (Wt stored N-major: Wt[n*K + k]).
//    Block = 4 waves = 128x64 tile; wave = 32(M) x 64(N), 8 accumulators.
//    B tile (64x32 f16, 4KB) staged in LDS once per block per K-step
//    (async copy) and shared by all 4 waves.
// ---------------------------------------------------------------------------
template <bool OUTF32>
__global__ void gemm_kernel(const _Float16* __restrict__ A,
                            const _Float16* __restrict__ Wt,
                            void* __restrict__ Cany,
                            int N, int K) {
  __shared__ _Float16 Blds[64 * 32];           // [n][k]

  const int wave = threadIdx.x >> 5;
  const int lane = threadIdx.x & 31;
  const int col  = lane & 15;
  const int hi   = lane >> 4;
  const int m0   = blockIdx.y * 128 + wave * 32;
  const int n0   = blockIdx.x * 64;

  v8f acc[2][4] = {};

  for (int kk = 0; kk < K; kk += 32) {
    __syncthreads();                           // previous tile fully consumed
    {
      // cooperative stage: 128 threads x 2 x B128 = 4KB
      const int n = threadIdx.x >> 1;
      const int c = (threadIdx.x & 1) * 16;
      const _Float16* src = Wt + (size_t)(n0 + n) * K + kk + c;
      _Float16*       dst = &Blds[n * 32 + c];
      stage_b128(src, dst);
      stage_b128(src + 8, dst + 8);
    }
    stage_wait();
    __syncthreads();

    // A fragments: lane holds row m=col; halves[0..7]=K c0.., [8..15]=c0+16..
    V16 a[2];
#pragma unroll
    for (int mi = 0; mi < 2; ++mi) {
      const _Float16* ap = A + (size_t)(m0 + mi * 16 + col) * K + kk + (hi ? 8 : 0);
      a[mi].h[0] = *(const v8h*)(ap);
      a[mi].h[1] = *(const v8h*)(ap + 16);
    }
#pragma unroll
    for (int j = 0; j < 4; ++j) {
      const _Float16* bp = &Blds[(j * 16 + col) * 32 + (hi ? 8 : 0)];
      V16 b;
      b.h[0] = *(const v8h*)(bp);
      b.h[1] = *(const v8h*)(bp + 16);
#pragma unroll
      for (int mi = 0; mi < 2; ++mi)
        acc[mi][j] = wmma_f16(a[mi].v, b.v, acc[mi][j]);
    }
  }

  // D layout: VGPR r -> (m = r + 8*hi, n = col)
#pragma unroll
  for (int mi = 0; mi < 2; ++mi)
#pragma unroll
    for (int j = 0; j < 4; ++j)
#pragma unroll
      for (int r = 0; r < 8; ++r) {
        const size_t idx = (size_t)(m0 + mi * 16 + r + 8 * hi) * N + n0 + j * 16 + col;
        if constexpr (OUTF32) ((float*)Cany)[idx]     = acc[mi][j][r];
        else                  ((_Float16*)Cany)[idx]  = (_Float16)acc[mi][j][r];
      }
}

// ---------------------------------------------------------------------------
// 3) Flash attention per (b, h, 64-query-row block). 4 waves / block,
//    each wave owns 16 query rows. Key loop step = 32 (one WMMA K-dim).
//    K tile staged row-major in LDS (async), V tile staged transposed.
// ---------------------------------------------------------------------------
__global__ void attn_kernel(const _Float16* __restrict__ Qp,
                            const _Float16* __restrict__ Kp,
                            const _Float16* __restrict__ Vp,
                            const int* __restrict__ valid_lens,
                            _Float16* __restrict__ Op) {
  __shared__ _Float16 Klds[32 * 64];      // [key][dh]   4 KB
  __shared__ _Float16 Vt[DH * 32];        // [dh][key]   4 KB
  __shared__ _Float16 Pt[4][16 * 32];     // per-wave P tile 4 KB

  const int wave = threadIdx.x >> 5;
  const int lane = threadIdx.x & 31;
  const int col  = lane & 15;
  const int hi   = lane >> 4;
  const int b    = blockIdx.z;
  const int h    = blockIdx.y;
  const int q0   = blockIdx.x * 64 + wave * 16;
  const int vl   = valid_lens[b];

  // Resident Q A-fragments (dh = 64 -> two K=32 fragments)
  V16 qf[2];
  {
    const _Float16* qb = Qp + (size_t)(b * S + q0 + col) * D + h * DH;
#pragma unroll
    for (int w = 0; w < 2; ++w) {
      const int c0 = w * 32 + (hi ? 8 : 0);
      qf[w].h[0] = *(const v8h*)(qb + c0);
      qf[w].h[1] = *(const v8h*)(qb + c0 + 16);
    }
  }

  float mstate[8], lstate[8];
#pragma unroll
  for (int r = 0; r < 8; ++r) { mstate[r] = -1e30f; lstate[r] = 0.0f; }
  v8f acc[4] = {};

  for (int kt = 0; kt < S; kt += 32) {
    __syncthreads();                      // previous K/V tiles consumed
    {
      // stage K rows (row-major) via async copy: 128 thr x 2 x B128 = 4KB
      const int key = threadIdx.x >> 2;
      const int c   = (threadIdx.x & 3) * 16;
      const _Float16* src = Kp + (size_t)(b * S + kt + key) * D + h * DH + c;
      _Float16*       dst = &Klds[key * 64 + c];
      stage_b128(src, dst);
      stage_b128(src + 8, dst + 8);
    }
    {
      // stage V transposed (manual scatter; transpose not expressible as async)
      const int k  = threadIdx.x & 31;
      const int c0 = (threadIdx.x >> 5) * 16;
      const _Float16* vrow = Vp + (size_t)(b * S + kt + k) * D + h * DH + c0;
      v8h va = *(const v8h*)(vrow);
      v8h vb = *(const v8h*)(vrow + 8);
#pragma unroll
      for (int i = 0; i < 8; ++i) Vt[(c0 + i) * 32 + k]     = va[i];
#pragma unroll
      for (int i = 0; i < 8; ++i) Vt[(c0 + 8 + i) * 32 + k] = vb[i];
      if (kt + 32 < S) __builtin_prefetch(vrow + (size_t)32 * D, 0, 0);
    }
    stage_wait();
    __syncthreads();

    // ---- preload all K^T B-fragments, then batch the 4 score WMMAs ----
    V16 kf[2][2];
#pragma unroll
    for (int nc = 0; nc < 2; ++nc)
#pragma unroll
      for (int w = 0; w < 2; ++w) {
        const _Float16* bp = &Klds[(nc * 16 + col) * 64 + w * 32 + (hi ? 8 : 0)];
        kf[nc][w].h[0] = *(const v8h*)(bp);
        kf[nc][w].h[1] = *(const v8h*)(bp + 16);
      }

    v8f st[2];
#pragma unroll
    for (int nc = 0; nc < 2; ++nc) {
      v8f s = {};
      s = wmma_f16(qf[0].v, kf[nc][0].v, s);
      s = wmma_f16(qf[1].v, kf[nc][1].v, s);
      const int key = kt + nc * 16 + col;
#pragma unroll
      for (int r = 0; r < 8; ++r)
        s[r] = (key >= vl) ? -1e6f : s[r] * 0.125f;   // scale then mask (ref order)
      st[nc] = s;
    }

    // ---- online softmax (row r lives across 16 lanes of a half-wave) ----
    float sf[8];
#pragma unroll
    for (int r = 0; r < 8; ++r) {
      float v = fmaxf(st[0][r], st[1][r]);
      v = fmaxf(v, __shfl_xor(v, 1, 32));
      v = fmaxf(v, __shfl_xor(v, 2, 32));
      v = fmaxf(v, __shfl_xor(v, 4, 32));
      v = fmaxf(v, __shfl_xor(v, 8, 32));
      const float mnew = fmaxf(mstate[r], v);
      sf[r] = __expf(mstate[r] - mnew);
      mstate[r] = mnew;
    }
#pragma unroll
    for (int nc = 0; nc < 2; ++nc)
#pragma unroll
      for (int r = 0; r < 8; ++r) {
        const float p = __expf(st[nc][r] - mstate[r]);
        st[nc][r] = p;
        Pt[wave][(r + 8 * hi) * 32 + nc * 16 + col] = (_Float16)p;  // D -> row-major
      }
#pragma unroll
    for (int r = 0; r < 8; ++r) {
      float sum = st[0][r] + st[1][r];
      sum += __shfl_xor(sum, 1, 32);
      sum += __shfl_xor(sum, 2, 32);
      sum += __shfl_xor(sum, 4, 32);
      sum += __shfl_xor(sum, 8, 32);
      lstate[r] = lstate[r] * sf[r] + sum;
    }
#pragma unroll
    for (int j = 0; j < 4; ++j)
#pragma unroll
      for (int r = 0; r < 8; ++r) acc[j][r] *= sf[r];

    __syncthreads();                       // order Pt scalar stores vs vector loads

    // ---- P @ V : A = P (16x32), B = V (32x16 per dh chunk) from Vt ----
    V16 pf;
    {
      const _Float16* pb = &Pt[wave][col * 32 + (hi ? 8 : 0)];
      pf.h[0] = *(const v8h*)(pb);
      pf.h[1] = *(const v8h*)(pb + 16);
    }
    V16 vf[4];
#pragma unroll
    for (int j = 0; j < 4; ++j) {
      const _Float16* vb2 = &Vt[(j * 16 + col) * 32 + (hi ? 8 : 0)];
      vf[j].h[0] = *(const v8h*)(vb2);
      vf[j].h[1] = *(const v8h*)(vb2 + 16);
    }
#pragma unroll
    for (int j = 0; j < 4; ++j)
      acc[j] = wmma_f16(pf.v, vf[j].v, acc[j]);
  }

  // ---- epilogue: normalize by l and store f16 into [B*S, D] layout ----
#pragma unroll
  for (int j = 0; j < 4; ++j)
#pragma unroll
    for (int r = 0; r < 8; ++r) {
      const float o = acc[j][r] / lstate[r];
      Op[(size_t)(b * S + q0 + r + 8 * hi) * D + h * DH + j * 16 + col] = (_Float16)o;
    }
}

// ---------------------------------------------------------------------------
// Host launcher
// ---------------------------------------------------------------------------
extern "C" void kernel_launch(void* const* d_in, const int* in_sizes, int n_in,
                              void* d_out, int out_size, void* d_ws, size_t ws_size,
                              hipStream_t stream) {
  (void)in_sizes; (void)n_in; (void)out_size; (void)ws_size;

  const float* q  = (const float*)d_in[0];
  const float* k  = (const float*)d_in[1];
  const float* v  = (const float*)d_in[2];
  const int*   vl = (const int*)d_in[3];
  const float* Wq = (const float*)d_in[4];
  const float* Wk = (const float*)d_in[5];
  const float* Wv = (const float*)d_in[6];
  const float* Wo = (const float*)d_in[7];
  float* out = (float*)d_out;

  _Float16* ws16 = (_Float16*)d_ws;
  const size_t DD = (size_t)D * D;    // 1M halves
  const size_t MD = (size_t)MM * D;   // 4M halves
  _Float16* Wqt = ws16 + 0 * DD;
  _Float16* Wkt = ws16 + 1 * DD;
  _Float16* Wvt = ws16 + 2 * DD;
  _Float16* Wot = ws16 + 3 * DD;
  _Float16* Qh  = ws16 + 4 * DD;          // converted activations
  _Float16* Kh  = Qh + MD;
  _Float16* Vh  = Kh + MD;
  _Float16* Qp  = Vh + MD;                // projected
  _Float16* Kp  = Qp + MD;
  _Float16* Vp  = Kp + MD;
  _Float16* Ao  = Vp + MD;                // attn output; total 64 MB

  // 1) one-time f16 conversion (weights transposed, activations flat)
  {
    dim3 blk(256), wg((unsigned)(DD / 256)), ag((unsigned)(MD / 256));
    wconv_kernel<<<wg, blk, 0, stream>>>(Wq, Wqt);
    wconv_kernel<<<wg, blk, 0, stream>>>(Wk, Wkt);
    wconv_kernel<<<wg, blk, 0, stream>>>(Wv, Wvt);
    wconv_kernel<<<wg, blk, 0, stream>>>(Wo, Wot);
    aconv_kernel<<<ag, blk, 0, stream>>>(q, Qh);
    aconv_kernel<<<ag, blk, 0, stream>>>(k, Kh);
    aconv_kernel<<<ag, blk, 0, stream>>>(v, Vh);
  }
  // 2) Q/K/V projections (f16 in -> f16 out)
  {
    dim3 blk(128), grd(D / 64, MM / 128);
    gemm_kernel<false><<<grd, blk, 0, stream>>>(Qh, Wqt, Qp, D, D);
    gemm_kernel<false><<<grd, blk, 0, stream>>>(Kh, Wkt, Kp, D, D);
    gemm_kernel<false><<<grd, blk, 0, stream>>>(Vh, Wvt, Vp, D, D);
  }
  // 3) masked flash attention
  {
    dim3 blk(128), grd(S / 64, H, BB);
    attn_kernel<<<grd, blk, 0, stream>>>(Qp, Kp, Vp, vl, Ao);
  }
  // 4) output projection (f16 in -> f32 d_out)
  {
    dim3 blk(128), grd(D / 64, MM / 128);
    gemm_kernel<true><<<grd, blk, 0, stream>>>(Ao, Wot, out, D, D);
  }
}
